// Model_63883343561259
// MI455X (gfx1250) — compile-verified, matrix operand order checked by reading, not run on hardware
//
#include <hip/hip_runtime.h>
#include <cstdint>
#include <cstddef>

#define EMB   64
#define BSZ   4
#define NPT   2048
#define KNN   12
#define NITER 2
#define MSEL  341
#define BN_INVF 0.9999950000374997f

typedef _Float16 v16h __attribute__((ext_vector_type(16)));
typedef _Float16 v8h  __attribute__((ext_vector_type(8)));
typedef float    v8f  __attribute__((ext_vector_type(8)));

__device__ __forceinline__ v8f wmma_f16(v16h a, v16h b, v8f c){
  // D = A(16x32 f16) * B(32x16 f16) + C(16x16 f32)
  return __builtin_amdgcn_wmma_f32_16x16x32_f16(false, a, false, b, (short)0, c, false, false);
}

// Load one 16x32 f16 operand tile. Works for A (row-major [M][K]) and for B when B is
// stored column-major as [N][K]: per documented layout, lane l holds row (l&15),
// VGPR j pairs K = (j<4?0:16) + (l>>4)*8 + ... => two contiguous 8-half (16B) chunks.
__device__ __forceinline__ v16h load_frag(const _Float16* base, int stride, int row0, int koff){
  int lane = threadIdx.x & 31;
  int r = lane & 15, h = lane >> 4;
  const _Float16* p = base + (size_t)(row0 + r) * stride + koff + h * 8;
  v8h lo = *(const v8h*)p;
  v8h hi = *(const v8h*)(p + 16);
  v16h f;
  #pragma unroll
  for (int i = 0; i < 8; ++i){ f[i] = lo[i]; f[8 + i] = hi[i]; }
  return f;
}

// order-preserving float<->uint encoding for atomic max (deterministic reduction)
__device__ __forceinline__ unsigned enc_f32(float f){
  unsigned u = __float_as_uint(f);
  return (u & 0x80000000u) ? ~u : (u | 0x80000000u);
}
__device__ __forceinline__ float dec_f32(unsigned e){
  unsigned u = (e & 0x80000000u) ? (e & 0x7fffffffu) : ~e;
  return __uint_as_float(u);
}

// ---------------- weight prep ----------------
__global__ __launch_bounds__(256) void k_convpad(const float* __restrict__ W,
                                                 _Float16* __restrict__ dst, int in_c){
  int t = blockIdx.x * 256 + threadIdx.x;
  if (t >= 64 * 64) return;
  int o = t >> 6, c = t & 63;
  dst[t] = (_Float16)((c < in_c) ? W[o * in_c + c] : 0.f);
}

__global__ __launch_bounds__(256) void k_init_act(const float* __restrict__ pts,
                                                  _Float16* __restrict__ act){
  int t = blockIdx.x * 256 + threadIdx.x;
  int c = t & 63;
  size_t pn = (size_t)(t >> 6);
  act[t] = (_Float16)((c < 3) ? pts[pn * 3 + c] : 0.f);
}

// ---------------- knn (top-12 by -||d||^2, self included, lower index wins ties) ----
__global__ __launch_bounds__(256) void k_knn(const float* __restrict__ pts,
                                             int* __restrict__ nnidx){
  __shared__ float sp[NPT * 3];
  int b = blockIdx.x >> 3;
  int n = ((blockIdx.x & 7) << 8) + threadIdx.x;
  for (int i = threadIdx.x; i < NPT * 3; i += 256) sp[i] = pts[(size_t)b * NPT * 3 + i];
  __syncthreads();
  float px = sp[n * 3 + 0], py = sp[n * 3 + 1], pz = sp[n * 3 + 2];
  float bv[KNN]; int bi[KNN];
  #pragma unroll
  for (int k = 0; k < KNN; ++k){ bv[k] = -3.0e38f; bi[k] = 0; }
  for (int m = 0; m < NPT; ++m){
    float dx = px - sp[m * 3 + 0], dy = py - sp[m * 3 + 1], dz = pz - sp[m * 3 + 2];
    float d = -(dx * dx + dy * dy + dz * dz);
    if (d > bv[KNN - 1]){
      int pos = KNN - 1;
      #pragma unroll
      for (int q = KNN - 1; q > 0; --q){
        if (bv[q - 1] < d){ bv[q] = bv[q - 1]; bi[q] = bi[q - 1]; pos = q - 1; }
      }
      bv[pos] = d; bi[pos] = m;
    }
  }
  #pragma unroll
  for (int k = 0; k < KNN; ++k) nnidx[((size_t)b * NPT + n) * KNN + k] = bi[k];
}

// ---------------- fused EdgeConv layer: gather-diff -> W1/bn/relu -> W2+bias -> max_k -> W3+bias
__global__ __launch_bounds__(128) void k_propagate(
    const _Float16* __restrict__ actin, _Float16* __restrict__ actout,
    const int* __restrict__ nnidx,
    const _Float16* __restrict__ w1h, const _Float16* __restrict__ w2h,
    const _Float16* __restrict__ w3h,
    const float* __restrict__ g1, const float* __restrict__ b1,
    const float* __restrict__ bb2, const float* __restrict__ bb3)
{
  __shared__ __align__(16) _Float16 Esh[192 * EMB];
  __shared__ __align__(16) _Float16 Hsh[192 * EMB];
  __shared__ __align__(16) _Float16 Xsh[16 * EMB];
  int tid = threadIdx.x;
  int b = blockIdx.y;
  int n0 = blockIdx.x * 16;
  // build E (16 points x 12 neighbors = 192 cols), stored [col][ch]
  for (int col = tid; col < 192; col += 128){
    int p = col / 12, kk = col - p * 12;
    int n = n0 + p;
    int j = nnidx[((size_t)b * NPT + n) * KNN + kk];
    const v8h* aj = (const v8h*)(actin + ((size_t)b * NPT + j) * EMB);
    const v8h* an = (const v8h*)(actin + ((size_t)b * NPT + n) * EMB);
    v8h* e = (v8h*)(Esh + col * EMB);
    #pragma unroll
    for (int q2 = 0; q2 < 8; ++q2) e[q2] = aj[q2] - an[q2];
  }
  __syncthreads();
  int lane = tid & 31, wave = tid >> 5;
  int n16 = lane & 15, half = lane >> 4;
  // stage 1: h = relu(bn(W1 * E))   (each wave owns 3 col-tiles; wave-local LDS traffic)
  for (int q = 0; q < 3; ++q){
    int col0 = (wave * 3 + q) * 16;
    for (int mt = 0; mt < 4; ++mt){
      v8f acc = {};
      #pragma unroll
      for (int kt = 0; kt < 2; ++kt){
        v16h a  = load_frag(w1h, EMB, mt * 16, kt * 32);
        v16h bm = load_frag(Esh + col0 * EMB, EMB, 0, kt * 32);
        acc = wmma_f16(a, bm, acc);
      }
      int obase = mt * 16 + half * 8;
      v8h hv;
      #pragma unroll
      for (int r = 0; r < 8; ++r){
        int o = obase + r;
        float v = acc[r] * (g1[o] * BN_INVF) + b1[o];
        hv[r] = (_Float16)(v > 0.f ? v : 0.f);
      }
      *(v8h*)(Hsh + (col0 + n16) * EMB + obase) = hv;
    }
  }
  // stage 2: h2 = W2 * h + bb2 (into Esh, wave-local)
  for (int q = 0; q < 3; ++q){
    int col0 = (wave * 3 + q) * 16;
    for (int mt = 0; mt < 4; ++mt){
      v8f acc = {};
      #pragma unroll
      for (int kt = 0; kt < 2; ++kt){
        v16h a  = load_frag(w2h, EMB, mt * 16, kt * 32);
        v16h bm = load_frag(Hsh + col0 * EMB, EMB, 0, kt * 32);
        acc = wmma_f16(a, bm, acc);
      }
      int obase = mt * 16 + half * 8;
      v8h hv;
      #pragma unroll
      for (int r = 0; r < 8; ++r) hv[r] = (_Float16)(acc[r] + bb2[obase + r]);
      *(v8h*)(Esh + (col0 + n16) * EMB + obase) = hv;
    }
  }
  __syncthreads();
  // max over K neighbors
  for (int i = tid; i < 16 * EMB; i += 128){
    int p = i >> 6, c = i & 63;
    float m = -3.0e38f;
    for (int kk = 0; kk < KNN; ++kk){
      float v = (float)Esh[(p * KNN + kk) * EMB + c];
      m = v > m ? v : m;
    }
    Xsh[p * EMB + c] = (_Float16)m;
  }
  __syncthreads();
  // stage 3: out = W3 * hmax + bb3  (single col-tile; wave 0)
  if (wave == 0){
    for (int mt = 0; mt < 4; ++mt){
      v8f acc = {};
      #pragma unroll
      for (int kt = 0; kt < 2; ++kt){
        v16h a  = load_frag(w3h, EMB, mt * 16, kt * 32);
        v16h bm = load_frag(Xsh, EMB, 0, kt * 32);
        acc = wmma_f16(a, bm, acc);
      }
      int obase = mt * 16 + half * 8;
      v8h ov;
      #pragma unroll
      for (int r = 0; r < 8; ++r) ov[r] = (_Float16)(acc[r] + bb3[obase + r]);
      *(v8h*)(actout + ((size_t)b * NPT + n0 + n16) * EMB + obase) = ov;
    }
  }
}

// ---------------- significance MLP (64->64->32->1), one thread per point --------
__global__ __launch_bounds__(256) void k_sig(
    const _Float16* __restrict__ emb,
    const float* __restrict__ w1, const float* __restrict__ g1, const float* __restrict__ b1,
    const float* __restrict__ w2, const float* __restrict__ g2, const float* __restrict__ b2,
    const float* __restrict__ w3, const float* __restrict__ bb3,
    float* __restrict__ out)
{
  __shared__ float sw1[64 * 64];
  __shared__ float sw2[32 * 64];
  __shared__ float sw3[32], sg1[64], sb1[64], sg2[32], sb2[32];
  int tid = threadIdx.x;
  for (int i = tid; i < 4096; i += 256) sw1[i] = w1[i];
  for (int i = tid; i < 2048; i += 256) sw2[i] = w2[i];
  if (tid < 64){ sg1[tid] = g1[tid] * BN_INVF; sb1[tid] = b1[tid]; }
  if (tid < 32){ sw3[tid] = w3[tid]; sg2[tid] = g2[tid] * BN_INVF; sb2[tid] = b2[tid]; }
  __syncthreads();
  int b = blockIdx.y;
  int n = blockIdx.x * 256 + tid;
  float e[64];
  const _Float16* er = emb + ((size_t)b * NPT + n) * EMB;
  for (int c = 0; c < 64; ++c) e[c] = (float)er[c];
  float h1[64];
  for (int o = 0; o < 64; ++o){
    float a = 0.f;
    for (int c = 0; c < 64; ++c) a += sw1[o * 64 + c] * e[c];
    a = a * sg1[o] + sb1[o];
    h1[o] = a > 0.f ? a : 0.f;
  }
  float acc = bb3[0];
  for (int o = 0; o < 32; ++o){
    float a = 0.f;
    for (int c = 0; c < 64; ++c) a += sw2[o * 64 + c] * h1[c];
    a = a * sg2[o] + sb2[o];
    acc += sw3[o] * (a > 0.f ? a : 0.f);
  }
  out[(size_t)b * NPT + n] = acc;
}

// ---------------- full in-LDS bitonic sort -> top-M indices (descending, idx tiebreak)
__global__ __launch_bounds__(1024) void k_topk(
    const float* __restrict__ sig_s, const float* __restrict__ sig_t,
    int* __restrict__ sel_s, int* __restrict__ sel_t)
{
  __shared__ float sv[NPT];
  __shared__ int   si[NPT];
  int tid = threadIdx.x;
  int b = blockIdx.x;
  const float* s = blockIdx.y ? sig_t : sig_s;
  int* o = blockIdx.y ? sel_t : sel_s;
  for (int i = tid; i < NPT; i += 1024){ sv[i] = s[(size_t)b * NPT + i]; si[i] = i; }
  __syncthreads();
  for (int k = 2; k <= NPT; k <<= 1){
    for (int j = k >> 1; j > 0; j >>= 1){
      for (int i = tid; i < NPT; i += 1024){
        int l = i ^ j;
        if (l > i){
          float vi = sv[i], vl = sv[l];
          int ii = si[i], il = si[l];
          bool pre = (vi > vl) || (vi == vl && ii < il);
          bool dir = ((i & k) == 0);
          if (dir ? !pre : pre){ sv[i] = vl; sv[l] = vi; si[i] = il; si[l] = ii; }
        }
      }
      __syncthreads();
    }
  }
  for (int i = tid; i < MSEL; i += 1024) o[(size_t)b * MSEL + i] = si[i];
}

// ---------------- gather selected points + embeddings; init R=I,t=0 ---------------
__global__ __launch_bounds__(128) void k_gather(
    const float* __restrict__ src, const float* __restrict__ tgt,
    const _Float16* __restrict__ emb_s, const _Float16* __restrict__ emb_t,
    const int* __restrict__ sel_s, const int* __restrict__ sel_t,
    float* __restrict__ srcs, float* __restrict__ tgts,
    _Float16* __restrict__ se, _Float16* __restrict__ te,
    float* __restrict__ Rt)
{
  int t = blockIdx.x * 128 + threadIdx.x;
  if (t < BSZ * 24){
    int b = t / 24, q = t - b * 24;
    float v = 0.f;
    if (q < 9) v = (q == 0 || q == 4 || q == 8) ? 1.f : 0.f;
    Rt[b * 24 + q] = v;
  }
  int total = 2 * BSZ * MSEL;
  if (t >= total) return;
  int which = t / (BSZ * MSEL);
  int r = t - which * (BSZ * MSEL);
  int b = r / MSEL, m = r - b * MSEL;
  const int* sel = which ? sel_t : sel_s;
  const float* pts = which ? tgt : src;
  const _Float16* em = which ? emb_t : emb_s;
  float* po = which ? tgts : srcs;
  _Float16* eo = which ? te : se;
  int idx = sel[(size_t)b * MSEL + m];
  for (int c = 0; c < 3; ++c)
    po[((size_t)b * MSEL + m) * 3 + c] = pts[((size_t)b * NPT + idx) * 3 + c];
  const v8h* erow = (const v8h*)(em + ((size_t)b * NPT + idx) * EMB);
  v8h* orow = (v8h*)(eo + ((size_t)b * MSEL + m) * EMB);
  #pragma unroll
  for (int q2 = 0; q2 < 8; ++q2) orow[q2] = erow[q2];
}

// ---------------- per-iteration prep: P = W1a*se, Q = W1b*te, zero wmax, f16 weights
__global__ __launch_bounds__(128) void k_iter_prep(
    const _Float16* __restrict__ se, const _Float16* __restrict__ te,
    const float* __restrict__ sm1w1,
    const float* __restrict__ sm1w2f, const float* __restrict__ sm2w1f,
    float* __restrict__ P, float* __restrict__ Q,
    unsigned* __restrict__ wmax, _Float16* __restrict__ iterWh)
{
  int t = blockIdx.x * 128 + threadIdx.x;
  if (t < 1024) iterWh[t] = (_Float16)sm1w2f[t];
  else if (t < 2048) iterWh[t] = (_Float16)sm2w1f[t - 1024];
  if (t >= BSZ * MSEL) return;
  int b = t / MSEL, i = t - b * MSEL;
  float ef[64], tf[64];
  const _Float16* sr = se + ((size_t)b * MSEL + i) * EMB;
  const _Float16* tr = te + ((size_t)b * MSEL + i) * EMB;
  for (int c = 0; c < 64; ++c){ ef[c] = (float)sr[c]; tf[c] = (float)tr[c]; }
  for (int o = 0; o < 32; ++o){
    float ap = 0.f, aq = 0.f;
    for (int c = 0; c < 64; ++c){
      ap += sm1w1[o * 132 + c]      * ef[c];
      aq += sm1w1[o * 132 + 64 + c] * tf[c];
    }
    size_t base = ((size_t)b * MSEL + i) * 32 + o;
    P[base] = ap; Q[base] = aq;
    wmax[base] = 0u;   // encoded -inf
  }
}

// ---------------- sim stage: rank-4 features + two 32x32 WMMA GEMMs over 16x16 pair tiles
__global__ __launch_bounds__(128) void k_sim(
    const float* __restrict__ srcs, const float* __restrict__ tgts,
    const float* __restrict__ P, const float* __restrict__ Q,
    const float* __restrict__ sm1w1, const float* __restrict__ sm1g1,
    const float* __restrict__ sm1b1, const float* __restrict__ sm1bb2,
    const _Float16* __restrict__ sm1w2h, const _Float16* __restrict__ sm2w1h,
    const float* __restrict__ sm2g1, const float* __restrict__ sm2b1,
    const float* __restrict__ sm2w2, const float* __restrict__ sm2bb2,
    unsigned* __restrict__ wmax, float* __restrict__ simout)
{
  __shared__ __align__(16) _Float16 Hsh[256 * 32];
  __shared__ __align__(16) _Float16 H2sh[256 * 32];
  __shared__ float s_wc[32 * 4], s_gs1[32], s_bs1[32], s_bb2a[32];
  __shared__ float s_gs2[32], s_bs2[32], s_w22[32];
  __shared__ float s_bb2b;
  int tid = threadIdx.x;
  int i0 = blockIdx.x * 16, j0 = blockIdx.y * 16, bz = blockIdx.z;
  if (tid < 32){
    s_gs1[tid] = sm1g1[tid] * BN_INVF; s_bs1[tid] = sm1b1[tid];
    s_bb2a[tid] = sm1bb2[tid];
    s_gs2[tid] = sm2g1[tid] * BN_INVF; s_bs2[tid] = sm2b1[tid];
    s_w22[tid] = sm2w2[tid];
    #pragma unroll
    for (int q = 0; q < 4; ++q) s_wc[tid * 4 + q] = sm1w1[tid * 132 + 128 + q];
    if (tid == 0) s_bb2b = sm2bb2[0];
  }
  __syncthreads();
  // h1 = relu(bn(P_i + Q_j + W1c * [dist; diffn]))   (col = i_local*16 + j_local)
  for (int col = tid; col < 256; col += 128){
    int il = col >> 4, jl = col & 15;
    int ii = i0 + il, jj = j0 + jl;
    if (ii < MSEL && jj < MSEL){
      const float* sp = srcs + ((size_t)bz * MSEL + ii) * 3;
      const float* tp = tgts + ((size_t)bz * MSEL + jj) * 3;
      float dx = sp[0] - tp[0], dy = sp[1] - tp[1], dz = sp[2] - tp[2];
      float dist = sqrtf(dx * dx + dy * dy + dz * dz);
      float inv = 1.f / (dist + 1e-8f);
      float f1 = dx * inv, f2 = dy * inv, f3 = dz * inv;
      const float* Pr = P + ((size_t)bz * MSEL + ii) * 32;
      const float* Qr = Q + ((size_t)bz * MSEL + jj) * 32;
      for (int o = 0; o < 32; ++o){
        float pre = Pr[o] + Qr[o] + s_wc[o * 4 + 0] * dist + s_wc[o * 4 + 1] * f1
                  + s_wc[o * 4 + 2] * f2 + s_wc[o * 4 + 3] * f3;
        float h = pre * s_gs1[o] + s_bs1[o];
        Hsh[col * 32 + o] = (_Float16)(h > 0.f ? h : 0.f);
      }
    } else {
      for (int o = 0; o < 32; ++o) Hsh[col * 32 + o] = (_Float16)0.f;
    }
  }
  __syncthreads();
  int lane = tid & 31, wave = tid >> 5;
  int n16 = lane & 15, half = lane >> 4;
  for (int q = 0; q < 4; ++q){
    int ct = wave * 4 + q;
    int col0 = ct * 16;
    int ii = i0 + ct, jj = j0 + n16;           // one col-tile <-> one i, lanes <-> j
    v16h bfrag = load_frag(Hsh + col0 * 32, 32, 0, 0);
    for (int mt = 0; mt < 2; ++mt){
      int obase = mt * 16 + half * 8;
      { // hA = sm1_w2 * h + bb2 -> lane-reduce max over j, then one atomic per (i,o)
        v16h a = load_frag(sm1w2h, 32, mt * 16, 0);
        v8f z = {};
        v8f acc = wmma_f16(a, bfrag, z);
        #pragma unroll
        for (int r = 0; r < 8; ++r){
          float v = acc[r] + s_bb2a[obase + r];
          if (jj >= MSEL) v = -3.0e38f;
          #pragma unroll
          for (int off = 1; off < 16; off <<= 1){
            float v2 = __shfl_xor(v, off, 32);
            v = v2 > v ? v2 : v;
          }
          if (n16 == 0 && ii < MSEL)
            atomicMax(&wmax[((size_t)bz * MSEL + ii) * 32 + obase + r], enc_f32(v));
        }
      }
      { // h2 = relu(bn(sm2_w1 * h))
        v16h a = load_frag(sm2w1h, 32, mt * 16, 0);
        v8f z = {};
        v8f acc = wmma_f16(a, bfrag, z);
        v8h hv;
        #pragma unroll
        for (int r = 0; r < 8; ++r){
          int o = obase + r;
          float v = acc[r] * s_gs2[o] + s_bs2[o];
          hv[r] = (_Float16)(v > 0.f ? v : 0.f);
        }
        *(v8h*)(H2sh + (col0 + n16) * 32 + obase) = hv;
      }
    }
  }
  __syncthreads();
  // sim2 = clip(sm2_w2 . h2 + bb2, +-20)
  for (int col = tid; col < 256; col += 128){
    int ii = i0 + (col >> 4), jj = j0 + (col & 15);
    if (ii < MSEL && jj < MSEL){
      float s = s_bb2b;
      for (int o = 0; o < 32; ++o) s += s_w22[o] * (float)H2sh[col * 32 + o];
      s = fminf(fmaxf(s, -20.f), 20.f);
      simout[((size_t)bz * MSEL + ii) * MSEL + jj] = s;
    }
  }
}

// ---------------- weight head: wf MLP + sigmoid + median threshold + normalize -----
__global__ __launch_bounds__(256) void k_wf(
    const unsigned* __restrict__ wmax,
    const float* __restrict__ wfw1, const float* __restrict__ wfg1, const float* __restrict__ wfb1,
    const float* __restrict__ wfw2, const float* __restrict__ wfbb2,
    float* __restrict__ wvec)
{
  __shared__ float sw[MSEL], sw2[MSEL];
  __shared__ float smed, ssum;
  int tid = threadIdx.x, b = blockIdx.x;
  for (int i = tid; i < MSEL; i += 256){
    float wm[32];
    for (int o = 0; o < 32; ++o) wm[o] = dec_f32(wmax[((size_t)b * MSEL + i) * 32 + o]);
    float acc = wfbb2[0];
    for (int o = 0; o < 32; ++o){
      float a = 0.f;
      for (int c = 0; c < 32; ++c) a += wfw1[o * 32 + c] * wm[c];
      a = a * (wfg1[o] * BN_INVF) + wfb1[o];
      acc += wfw2[o] * (a > 0.f ? a : 0.f);
    }
    sw[i] = 1.f / (1.f + expf(-acc));
  }
  __syncthreads();
  for (int i = tid; i < MSEL; i += 256){      // rank-based median (sorted index 170)
    float v = sw[i];
    int clt = 0, cle = 0;
    for (int j = 0; j < MSEL; ++j){
      float u = sw[j];
      clt += (u < v); cle += (u <= v);
    }
    if (clt <= (MSEL - 1) / 2 && (MSEL - 1) / 2 < cle) smed = v;
  }
  __syncthreads();
  for (int i = tid; i < MSEL; i += 256){
    float v = sw[i];
    sw2[i] = (v >= smed) ? v : 0.f;
  }
  __syncthreads();
  if (tid == 0){                               // serial sum: deterministic
    float s = 0.f;
    for (int j = 0; j < MSEL; ++j) s += sw2[j];
    ssum = s + 1e-8f;
  }
  __syncthreads();
  for (int i = tid; i < MSEL; i += 256) wvec[(size_t)b * MSEL + i] = sw2[i] / ssum;
}

// ---------------- row argmax -> correspondence gather -----------------------------
__global__ __launch_bounds__(128) void k_corr(const float* __restrict__ sim,
                                              const float* __restrict__ tgts,
                                              float* __restrict__ corr)
{
  int t = blockIdx.x * 128 + threadIdx.x;
  if (t >= BSZ * MSEL) return;
  int b = t / MSEL, i = t - b * MSEL;
  const float* row = sim + ((size_t)b * MSEL + i) * MSEL;
  float best = row[0]; int jb = 0;
  for (int j = 1; j < MSEL; ++j){
    float v = row[j];
    if (v > best){ best = v; jb = j; }
  }
  for (int c = 0; c < 3; ++c)
    corr[(size_t)t * 3 + c] = tgts[((size_t)b * MSEL + jb) * 3 + c];
}

// ---------------- weighted Kabsch via 3x3 Jacobi SVD ------------------------------
__device__ __forceinline__ float det3(const float* A){
  return A[0]*(A[4]*A[8]-A[5]*A[7]) - A[1]*(A[3]*A[8]-A[5]*A[6]) + A[2]*(A[3]*A[7]-A[4]*A[6]);
}

__global__ __launch_bounds__(32) void k_svd(const float* __restrict__ srcs,
                                            const float* __restrict__ corr,
                                            const float* __restrict__ wvec,
                                            float* __restrict__ Rt)
{
  if (threadIdx.x != 0) return;
  int b = blockIdx.x;
  const float* S = srcs + (size_t)b * MSEL * 3;
  const float* C = corr + (size_t)b * MSEL * 3;
  const float* W = wvec + (size_t)b * MSEL;
  float ms[3] = {0, 0, 0}, mc[3] = {0, 0, 0};
  for (int n = 0; n < MSEL; ++n){
    float wn = W[n];
    for (int c = 0; c < 3; ++c){ ms[c] += wn * S[n * 3 + c]; mc[c] += wn * C[n * 3 + c]; }
  }
  float H[9] = {0, 0, 0, 0, 0, 0, 0, 0, 0};
  for (int n = 0; n < MSEL; ++n){
    float wn = W[n];
    float sx[3], cx[3];
    for (int c = 0; c < 3; ++c){ sx[c] = S[n * 3 + c] - ms[c]; cx[c] = C[n * 3 + c] - mc[c]; }
    for (int i2 = 0; i2 < 3; ++i2)
      for (int j2 = 0; j2 < 3; ++j2)
        H[i2 * 3 + j2] += wn * sx[i2] * cx[j2];
  }
  float A[9], V[9] = {1, 0, 0, 0, 1, 0, 0, 0, 1};
  for (int i2 = 0; i2 < 3; ++i2)
    for (int j2 = 0; j2 < 3; ++j2)
      A[i2 * 3 + j2] = H[0 + i2] * H[0 + j2] + H[3 + i2] * H[3 + j2] + H[6 + i2] * H[6 + j2];
  for (int sweep = 0; sweep < 24; ++sweep){
    int s3 = sweep % 3;
    int p = (s3 == 2) ? 1 : 0;
    int q = (s3 == 0) ? 1 : 2;
    float apq = A[p * 3 + q];
    if (fabsf(apq) < 1e-30f) continue;
    float tau = (A[q * 3 + q] - A[p * 3 + p]) / (2.f * apq);
    float tnum = ((tau >= 0.f) ? 1.f : -1.f) / (fabsf(tau) + sqrtf(1.f + tau * tau));
    float c1 = rsqrtf(1.f + tnum * tnum);
    float s1 = tnum * c1;
    for (int k = 0; k < 3; ++k){
      float akp = A[k * 3 + p], akq = A[k * 3 + q];
      A[k * 3 + p] = c1 * akp - s1 * akq;
      A[k * 3 + q] = s1 * akp + c1 * akq;
    }
    for (int k = 0; k < 3; ++k){
      float apk = A[p * 3 + k], aqk = A[q * 3 + k];
      A[p * 3 + k] = c1 * apk - s1 * aqk;
      A[q * 3 + k] = s1 * apk + c1 * aqk;
    }
    for (int k = 0; k < 3; ++k){
      float vkp = V[k * 3 + p], vkq = V[k * 3 + q];
      V[k * 3 + p] = c1 * vkp - s1 * vkq;
      V[k * 3 + q] = s1 * vkp + c1 * vkq;
    }
  }
  float lam[3] = {A[0], A[4], A[8]};
  int ord[3] = {0, 1, 2};
  for (int a2 = 0; a2 < 2; ++a2)
    for (int b2 = a2 + 1; b2 < 3; ++b2)
      if (lam[ord[b2]] > lam[ord[a2]]){ int tp = ord[a2]; ord[a2] = ord[b2]; ord[b2] = tp; }
  float Vs[9];
  for (int k = 0; k < 3; ++k)
    for (int i2 = 0; i2 < 3; ++i2)
      Vs[i2 * 3 + k] = V[i2 * 3 + ord[k]];
  float U[9];
  for (int k = 0; k < 2; ++k){
    float u[3];
    for (int i2 = 0; i2 < 3; ++i2)
      u[i2] = H[i2 * 3 + 0] * Vs[0 * 3 + k] + H[i2 * 3 + 1] * Vs[1 * 3 + k] + H[i2 * 3 + 2] * Vs[2 * 3 + k];
    float nn = sqrtf(u[0] * u[0] + u[1] * u[1] + u[2] * u[2]);
    if (nn > 1e-20f){
      float inv = 1.f / nn;
      for (int i2 = 0; i2 < 3; ++i2) U[i2 * 3 + k] = u[i2] * inv;
    } else {
      for (int i2 = 0; i2 < 3; ++i2) U[i2 * 3 + k] = (i2 == k) ? 1.f : 0.f;
    }
  }
  U[0 * 3 + 2] = U[1 * 3 + 0] * U[2 * 3 + 1] - U[2 * 3 + 0] * U[1 * 3 + 1];
  U[1 * 3 + 2] = U[2 * 3 + 0] * U[0 * 3 + 1] - U[0 * 3 + 0] * U[2 * 3 + 1];
  U[2 * 3 + 2] = U[0 * 3 + 0] * U[1 * 3 + 1] - U[1 * 3 + 0] * U[0 * 3 + 1];
  float d = det3(Vs) * det3(U);
  float Rab[9];
  for (int i2 = 0; i2 < 3; ++i2)
    for (int j2 = 0; j2 < 3; ++j2)
      Rab[i2 * 3 + j2] = Vs[i2 * 3 + 0] * U[j2 * 3 + 0] + Vs[i2 * 3 + 1] * U[j2 * 3 + 1]
                       + d * Vs[i2 * 3 + 2] * U[j2 * 3 + 2];
  float tab[3];
  for (int i2 = 0; i2 < 3; ++i2)
    tab[i2] = mc[i2] - (Rab[i2 * 3 + 0] * ms[0] + Rab[i2 * 3 + 1] * ms[1] + Rab[i2 * 3 + 2] * ms[2]);
  float* G = Rt + b * 24;
  float Rold[9], told[3];
  for (int k = 0; k < 9; ++k) Rold[k] = G[k];
  for (int k = 0; k < 3; ++k) told[k] = G[9 + k];
  for (int i2 = 0; i2 < 3; ++i2){
    for (int j2 = 0; j2 < 3; ++j2)
      G[i2 * 3 + j2] = Rab[i2 * 3 + 0] * Rold[0 * 3 + j2] + Rab[i2 * 3 + 1] * Rold[1 * 3 + j2]
                     + Rab[i2 * 3 + 2] * Rold[2 * 3 + j2];
    G[9 + i2] = Rab[i2 * 3 + 0] * told[0] + Rab[i2 * 3 + 1] * told[1]
              + Rab[i2 * 3 + 2] * told[2] + tab[i2];
  }
  for (int k = 0; k < 9; ++k) G[12 + k] = Rab[k];
  for (int k = 0; k < 3; ++k) G[21 + k] = tab[k];
}

__global__ __launch_bounds__(128) void k_xform(float* __restrict__ srcs,
                                               const float* __restrict__ Rt){
  int t = blockIdx.x * 128 + threadIdx.x;
  if (t >= BSZ * MSEL) return;
  int b = t / MSEL;
  const float* G = Rt + b * 24;
  float* p = srcs + (size_t)t * 3;
  float x = p[0], y = p[1], z = p[2];
  p[0] = G[12 + 0] * x + G[12 + 1] * y + G[12 + 2] * z + G[21];
  p[1] = G[12 + 3] * x + G[12 + 4] * y + G[12 + 5] * z + G[22];
  p[2] = G[12 + 6] * x + G[12 + 7] * y + G[12 + 8] * z + G[23];
}

__global__ __launch_bounds__(64) void k_final(const float* __restrict__ Rt,
                                              float* __restrict__ outT){
  int t = threadIdx.x;
  if (t >= BSZ * 16) return;
  int b = t >> 4, rc = t & 15, r = rc >> 2, c = rc & 3;
  const float* G = Rt + b * 24;
  float v;
  if (r < 3) v = (c < 3) ? G[r * 3 + c] : G[9 + r];
  else v = (c == 3) ? 1.f : 0.f;
  outT[t] = v;
}

// =================================================================================
extern "C" void kernel_launch(void* const* d_in, const int* in_sizes, int n_in,
                              void* d_out, int out_size, void* d_ws, size_t ws_size,
                              hipStream_t stream)
{
  (void)in_sizes; (void)n_in; (void)out_size; (void)ws_size;
  // input order (insertion order of setup_inputs and nested dicts):
  // 0:src 1:tgt, per prop layer l: 2+7l + {w1,g1,b1,w2,bb2,w3,bb3},
  // sig @37: {w1,g1,b1,w2,g2,b2,w3,bb3}, iters @45+15*it:
  // {sm1_w1,sm1_g1,sm1_b1,sm1_w2,sm1_bb2,sm2_w1,sm2_g1,sm2_b1,sm2_w2,sm2_bb2,
  //  wf_w1,wf_g1,wf_b1,wf_w2,wf_bb2}
  const float* src = (const float*)d_in[0];
  const float* tgt = (const float*)d_in[1];

  char* wsp = (char*)d_ws;
  auto take = [&](size_t bytes) -> void* {
    void* p = (void*)wsp;
    wsp += (bytes + 255) & ~(size_t)255;
    return p;
  };
  _Float16* wprop  = (_Float16*)take((size_t)5 * 3 * 64 * 64 * sizeof(_Float16));
  int* knn_s       = (int*)take((size_t)BSZ * NPT * KNN * sizeof(int));
  int* knn_t       = (int*)take((size_t)BSZ * NPT * KNN * sizeof(int));
  _Float16* actA   = (_Float16*)take((size_t)BSZ * NPT * EMB * sizeof(_Float16));
  _Float16* actB   = (_Float16*)take((size_t)BSZ * NPT * EMB * sizeof(_Float16));
  _Float16* emb_s  = (_Float16*)take((size_t)BSZ * NPT * EMB * sizeof(_Float16));
  _Float16* emb_t  = (_Float16*)take((size_t)BSZ * NPT * EMB * sizeof(_Float16));
  float* sig_s     = (float*)take((size_t)BSZ * NPT * sizeof(float));
  float* sig_t     = (float*)take((size_t)BSZ * NPT * sizeof(float));
  int* sel_s       = (int*)take((size_t)BSZ * MSEL * sizeof(int));
  int* sel_t       = (int*)take((size_t)BSZ * MSEL * sizeof(int));
  float* srcs      = (float*)take((size_t)BSZ * MSEL * 3 * sizeof(float));
  float* tgts      = (float*)take((size_t)BSZ * MSEL * 3 * sizeof(float));
  _Float16* se     = (_Float16*)take((size_t)BSZ * MSEL * EMB * sizeof(_Float16));
  _Float16* te     = (_Float16*)take((size_t)BSZ * MSEL * EMB * sizeof(_Float16));
  float* Pb        = (float*)take((size_t)BSZ * MSEL * 32 * sizeof(float));
  float* Qb        = (float*)take((size_t)BSZ * MSEL * 32 * sizeof(float));
  unsigned* wmax   = (unsigned*)take((size_t)BSZ * MSEL * 32 * sizeof(unsigned));
  float* wvec      = (float*)take((size_t)BSZ * MSEL * sizeof(float));
  _Float16* iterWh = (_Float16*)take((size_t)2048 * sizeof(_Float16));
  float* corr      = (float*)take((size_t)BSZ * MSEL * 3 * sizeof(float));
  float* Rt        = (float*)take((size_t)BSZ * 24 * sizeof(float));

  // pad + convert EdgeConv weights to f16 (uniform 64x64 tiles)
  for (int l = 0; l < 5; ++l){
    int in_c = (l == 0) ? 3 : 64;
    k_convpad<<<16, 256, 0, stream>>>((const float*)d_in[2 + l * 7 + 0], wprop + (l * 3 + 0) * 4096, in_c);
    k_convpad<<<16, 256, 0, stream>>>((const float*)d_in[2 + l * 7 + 3], wprop + (l * 3 + 1) * 4096, 64);
    k_convpad<<<16, 256, 0, stream>>>((const float*)d_in[2 + l * 7 + 5], wprop + (l * 3 + 2) * 4096, 64);
  }

  const float* pts2[2] = {src, tgt};
  int* nn2[2]          = {knn_s, knn_t};
  _Float16* emb2[2]    = {emb_s, emb_t};
  float* sig2[2]       = {sig_s, sig_t};
  for (int w = 0; w < 2; ++w){
    k_knn<<<BSZ * 8, 256, 0, stream>>>(pts2[w], nn2[w]);
    k_init_act<<<(BSZ * NPT * EMB) / 256, 256, 0, stream>>>(pts2[w], actA);
    const _Float16* cur = actA;
    for (int l = 0; l < 5; ++l){
      _Float16* outp = (l == 4) ? emb2[w] : ((l & 1) ? actA : actB);
      k_propagate<<<dim3(NPT / 16, BSZ), 128, 0, stream>>>(
          cur, outp, nn2[w],
          wprop + (l * 3 + 0) * 4096, wprop + (l * 3 + 1) * 4096, wprop + (l * 3 + 2) * 4096,
          (const float*)d_in[2 + l * 7 + 1], (const float*)d_in[2 + l * 7 + 2],
          (const float*)d_in[2 + l * 7 + 4], (const float*)d_in[2 + l * 7 + 6]);
      cur = outp;
    }
    k_sig<<<dim3(NPT / 256, BSZ), 256, 0, stream>>>(emb2[w],
        (const float*)d_in[37], (const float*)d_in[38], (const float*)d_in[39],
        (const float*)d_in[40], (const float*)d_in[41], (const float*)d_in[42],
        (const float*)d_in[43], (const float*)d_in[44], sig2[w]);
  }
  k_topk<<<dim3(BSZ, 2), 1024, 0, stream>>>(sig_s, sig_t, sel_s, sel_t);
  {
    int total = 2 * BSZ * MSEL;
    k_gather<<<(total + 127) / 128, 128, 0, stream>>>(src, tgt, emb_s, emb_t,
                                                      sel_s, sel_t, srcs, tgts, se, te, Rt);
  }
  float* simout = (float*)d_out + BSZ * 16;   // output: [T (B,4,4) | sim_last (B,M,M)]
  const int TIL = (MSEL + 15) / 16;
  for (int it = 0; it < NITER; ++it){
    int base = 45 + it * 15;
    const float* sm1w1  = (const float*)d_in[base + 0];
    const float* sm1g1  = (const float*)d_in[base + 1];
    const float* sm1b1  = (const float*)d_in[base + 2];
    const float* sm1w2  = (const float*)d_in[base + 3];
    const float* sm1bb2 = (const float*)d_in[base + 4];
    const float* sm2w1  = (const float*)d_in[base + 5];
    const float* sm2g1  = (const float*)d_in[base + 6];
    const float* sm2b1  = (const float*)d_in[base + 7];
    const float* sm2w2  = (const float*)d_in[base + 8];
    const float* sm2bb2 = (const float*)d_in[base + 9];
    const float* wfw1   = (const float*)d_in[base + 10];
    const float* wfg1   = (const float*)d_in[base + 11];
    const float* wfb1   = (const float*)d_in[base + 12];
    const float* wfw2   = (const float*)d_in[base + 13];
    const float* wfbb2  = (const float*)d_in[base + 14];
    k_iter_prep<<<16, 128, 0, stream>>>(se, te, sm1w1, sm1w2, sm2w1, Pb, Qb, wmax, iterWh);
    k_sim<<<dim3(TIL, TIL, BSZ), 128, 0, stream>>>(srcs, tgts, Pb, Qb,
        sm1w1, sm1g1, sm1b1, sm1bb2,
        iterWh, iterWh + 1024,
        sm2g1, sm2b1, sm2w2, sm2bb2, wmax, simout);
    k_wf<<<BSZ, 256, 0, stream>>>(wmax, wfw1, wfg1, wfb1, wfw2, wfbb2, wvec);
    k_corr<<<(BSZ * MSEL + 127) / 128, 128, 0, stream>>>(simout, tgts, corr);
    k_svd<<<BSZ, 32, 0, stream>>>(srcs, corr, wvec, Rt);
    k_xform<<<(BSZ * MSEL + 127) / 128, 128, 0, stream>>>(srcs, Rt);
  }
  k_final<<<1, 64, 0, stream>>>(Rt, (float*)d_out);
}
